// MultiHeadAttention_33835752358306
// MI455X (gfx1250) — compile-verified, matrix operand order checked by reading, not run on hardware
//
#include <hip/hip_runtime.h>
#include <cstdint>
#include <cstddef>

// ---------------------------------------------------------------------------
// CDNA5 (gfx1250) bf16 WMMA multi-head attention forward.
//  - all matmuls: v_wmma_f32_16x16x32_bf16 (wave32, 16x16 tiles, K=32)
//  - Q tile + out-proj A tile: async global->LDS (ASYNCcnt)
//  - K tile: Tensor Data Mover tensor_load_to_lds (TENSORcnt) with LDS padding
// ---------------------------------------------------------------------------

typedef __attribute__((ext_vector_type(16))) __bf16 v16bf;
typedef __attribute__((ext_vector_type(8)))  float  v8f;
typedef __attribute__((ext_vector_type(4)))  int    v4i;

union Frag32 {
    uint4 q[2];   // 32 bytes = 16 bf16 per lane
    v16bf v;
};
static_assert(sizeof(Frag32) == 32, "frag size");

__device__ __forceinline__ unsigned short f2bf(float f) {
    unsigned int u = __builtin_bit_cast(unsigned int, f);
    u += 0x7FFFu + ((u >> 16) & 1u);          // round-to-nearest-even
    return (unsigned short)(u >> 16);
}

// packed 2xf32 -> 2xbf16 (v_cvt_pk_bf16_f32 when available)
__device__ __forceinline__ unsigned int f2bf_pk(float lo, float hi) {
#if defined(__gfx1250__) && __has_builtin(__builtin_amdgcn_cvt_pk_bf16_f32)
    typedef __attribute__((ext_vector_type(2))) __bf16 v2bf;
    v2bf r = __builtin_amdgcn_cvt_pk_bf16_f32(lo, hi);
    return __builtin_bit_cast(unsigned int, r);
#else
    return (unsigned int)f2bf(lo) | ((unsigned int)f2bf(hi) << 16);
#endif
}

__device__ __forceinline__ v8f wmma_bf16(const Frag32& a, const Frag32& b, v8f c) {
    // (neg_a, A, neg_b, B, c_mod, C, reuse_a, reuse_b)
    return __builtin_amdgcn_wmma_f32_16x16x32_bf16(false, a.v, false, b.v,
                                                   (short)0, c, false, false);
}

// ---------------------------------------------------------------------------
// Async global->LDS 16-byte copy (per-lane), ASYNCcnt-tracked.
// ---------------------------------------------------------------------------
#if defined(__gfx1250__) && __has_builtin(__builtin_amdgcn_global_load_async_to_lds_b128)
#define USE_ASYNC_LDS 1
#else
#define USE_ASYNC_LDS 0
#endif

__device__ __forceinline__ void copy_b128_g2l(unsigned short* ldst,
                                              const unsigned short* gsrc) {
#if USE_ASYNC_LDS
    __builtin_amdgcn_global_load_async_to_lds_b128(
        (__attribute__((address_space(1))) v4i*)gsrc,
        (__attribute__((address_space(3))) v4i*)ldst,
        0, 0);
#else
    *(uint4*)ldst = *(const uint4*)gsrc;
#endif
}

__device__ __forceinline__ void wait_async0() {
#if USE_ASYNC_LDS
#if __has_builtin(__builtin_amdgcn_s_wait_asynccnt)
    __builtin_amdgcn_s_wait_asynccnt(0);
#else
    asm volatile("s_wait_asynccnt 0x0" ::: "memory");
#endif
#endif
}

// ---------------------------------------------------------------------------
// Tensor Data Mover: 2D bf16 tile load global->LDS with LDS row padding.
//   tile_x elems per row, tile_y rows, global row stride row_stride_elems,
//   pad_interval_code / pad_amount_code per D# group1 encoding.
// Issued by one wave; tracked by TENSORcnt.
// ---------------------------------------------------------------------------
#if defined(__gfx1250__)
__device__ __forceinline__ unsigned int lds_addr_of(void* p) {
    return (unsigned int)(unsigned long long)(__attribute__((address_space(3))) char*)p;
}

__device__ __forceinline__ void tdm_load_2d_bf16(
    unsigned int lds_off, const void* gptr,
    unsigned int tile_x, unsigned int tile_y,
    unsigned int row_stride_elems,
    unsigned int pad_interval_code, unsigned int pad_amount_code)
{
    typedef __attribute__((ext_vector_type(4))) unsigned int su4;
    typedef __attribute__((ext_vector_type(8))) unsigned int su8;
    unsigned long long ga = (unsigned long long)gptr;
    // D# group 0: count=1 | lds_addr | global_addr[56:0] | type=2
    su4 g0 = { 1u,
               lds_off,
               (unsigned int)ga,
               (unsigned int)((ga >> 32) & 0x01FFFFFFu) | (2u << 30) };
    // D# group 1: data_size=2B, pad_enable, pad codes, dims/strides
    su8 g1 = { (1u << 16) | (1u << 20) | (pad_interval_code << 22) | (pad_amount_code << 25),
               tile_x << 16,          // tensor_dim0[15:0]  @ bits 63:48
               tile_y << 16,          // tensor_dim1[15:0]  @ bits 95:80
               tile_x << 16,          // tile_dim0          @ bits 127:112
               tile_y,                // tile_dim1          @ bits 143:128
               row_stride_elems,      // tensor_dim0_stride[31:0]
               0u, 0u };
    asm volatile("tensor_load_to_lds %0, %1" :: "s"(g0), "s"(g1) : "memory");
}
#endif

// ---------------------------------------------------------------------------
// Generic tiled GEMM: C[M,N] = A[M,K] * B[K,N] + bias[N]
//   A: f32 (A_BF16=false) or bf16 (A_BF16=true), row-major
//   B: f32 row-major, converted to bf16 while staging (stored transposed)
//   C: bf16 (OUT_BF16=true) or f32
// Block tile 128x128, BK=32, 256 threads (8 waves), wave tile 32x64.
// ---------------------------------------------------------------------------
template<bool A_BF16, bool OUT_BF16>
__global__ __launch_bounds__(256) void gemm_bias_kernel(
    const void* __restrict__ Ap, const float* __restrict__ Bw,
    const float* __restrict__ bias, void* __restrict__ Cp,
    int M, int N, int K)
{
    __shared__ alignas(16) unsigned short As [128][40];  // 80B row stride (16B aligned)
    __shared__ alignas(16) unsigned short BsT[128][40];  // [n][k], transposed B tile

    const int tid  = threadIdx.x;
    const int lane = tid & 31;
    const int wave = tid >> 5;
    const int wr   = wave >> 1;     // wave row 0..3  (32 rows each)
    const int wc   = wave & 1;      // wave col 0..1  (64 cols each)
    const int mg   = lane & 15;
    const int g    = lane >> 4;

    const int bm = blockIdx.y * 128;
    const int bn = blockIdx.x * 128;

    v8f zero = {0,0,0,0,0,0,0,0};
    v8f acc[2][4];
    #pragma unroll
    for (int i = 0; i < 2; ++i)
        #pragma unroll
        for (int j = 0; j < 4; ++j) acc[i][j] = zero;

    for (int kt = 0; kt < K; kt += 32) {
        __syncthreads();   // protect LDS from previous iteration's readers

        // ---- stage A tile (128 x 32) as bf16 ----
        if (A_BF16) {
            const unsigned short* A = (const unsigned short*)Ap;
            #pragma unroll
            for (int j = 0; j < 2; ++j) {
                int lin = tid + j * 256;        // 512 chunks of 8 bf16
                int r   = lin >> 2;
                int c8  = lin & 3;
                copy_b128_g2l(&As[r][c8 * 8],
                              A + (size_t)(bm + r) * K + kt + c8 * 8);
            }
        } else {
            const float* A = (const float*)Ap;
            #pragma unroll
            for (int j = 0; j < 4; ++j) {
                int lin = tid + j * 256;        // 1024 chunks of 4 f32
                int r   = lin >> 3;
                int c4  = lin & 7;
                float4 d = *(const float4*)(A + (size_t)(bm + r) * K + kt + c4 * 4);
                uint2 p;
                p.x = f2bf_pk(d.x, d.y);
                p.y = f2bf_pk(d.z, d.w);
                *(uint2*)&As[r][c4 * 4] = p;
            }
        }

        // ---- stage B tile (32 x 128) transposed -> BsT[n][k], bf16 ----
        #pragma unroll
        for (int j = 0; j < 4; ++j) {
            int lin = tid + j * 256;            // 1024 chunks of 4 f32
            int k   = lin >> 5;
            int n4  = lin & 31;
            float4 d = *(const float4*)(Bw + (size_t)(kt + k) * N + bn + n4 * 4);
            BsT[n4 * 4 + 0][k] = f2bf(d.x);
            BsT[n4 * 4 + 1][k] = f2bf(d.y);
            BsT[n4 * 4 + 2][k] = f2bf(d.z);
            BsT[n4 * 4 + 3][k] = f2bf(d.w);
        }
        wait_async0();
        __syncthreads();

        // ---- WMMA: wave computes 2x4 tiles of 16x16 ----
        Frag32 a[2], b;
        #pragma unroll
        for (int ti = 0; ti < 2; ++ti) {
            int row = wr * 32 + ti * 16 + mg;
            a[ti].q[0] = *(const uint4*)&As[row][g * 8];
            a[ti].q[1] = *(const uint4*)&As[row][16 + g * 8];
        }
        #pragma unroll
        for (int tj = 0; tj < 4; ++tj) {
            int col = wc * 64 + tj * 16 + mg;
            b.q[0] = *(const uint4*)&BsT[col][g * 8];
            b.q[1] = *(const uint4*)&BsT[col][16 + g * 8];
            #pragma unroll
            for (int ti = 0; ti < 2; ++ti)
                acc[ti][tj] = wmma_bf16(a[ti], b, acc[ti][tj]);
        }
    }

    // ---- epilogue: bias + store ----
    #pragma unroll
    for (int tj = 0; tj < 4; ++tj) {
        int col = bn + wc * 64 + tj * 16 + mg;
        float bv = bias[col];
        #pragma unroll
        for (int ti = 0; ti < 2; ++ti) {
            #pragma unroll
            for (int r = 0; r < 8; ++r) {
                int row = bm + wr * 32 + ti * 16 + r + 8 * g;
                float v = acc[ti][tj][r] + bv;
                if (OUT_BF16)
                    ((unsigned short*)Cp)[(size_t)row * N + col] = f2bf(v);
                else
                    ((float*)Cp)[(size_t)row * N + col] = v;
            }
        }
    }
}

// ---------------------------------------------------------------------------
// Flash attention: one block = 128 query rows of one (batch, head).
// 8 waves, each owns 16 query rows. Online softmax, bf16 WMMA for QK^T & PV.
// qkv layout: [b, s, 3*1024] bf16 (Q @ +0, K @ +1024, V @ +2048).
// ---------------------------------------------------------------------------
__global__ __launch_bounds__(256) void attn_kernel(
    const unsigned short* __restrict__ qkv, unsigned short* __restrict__ Obuf)
{
    constexpr int S = 2048, HD = 64, C3 = 3072, E = 1024;

    __shared__ alignas(16) unsigned short Qs[128][72];     // query tile
    __shared__ alignas(16) unsigned short Ksh[64][72];     // K tile (row-major == B^T layout)
    __shared__ alignas(16) unsigned short Vt[64][72];      // V tile transposed [d][kv]
    __shared__ alignas(16) unsigned short Ps[8][16][72];   // per-wave P tile (16 x 64)

    const int tid  = threadIdx.x;
    const int lane = tid & 31;
    const int wave = tid >> 5;
    const int mg   = lane & 15;
    const int g    = lane >> 4;
    const int b    = blockIdx.y >> 4;
    const int h    = blockIdx.y & 15;
    const int q0   = blockIdx.x * 128;
    const unsigned short* base = qkv + (size_t)b * S * C3;

    // ---- stage Q (128 x 64), async global->LDS ----
    #pragma unroll
    for (int j = 0; j < 4; ++j) {
        int lin = tid + j * 256;                 // 1024 chunks of 8
        int r = lin >> 3, c8 = lin & 7;
        copy_b128_g2l(&Qs[r][c8 * 8],
                      base + (size_t)(q0 + r) * C3 + h * HD + c8 * 8);
    }
    wait_async0();
    __syncthreads();

    // per-wave Q fragments (constant over the whole kv loop)
    Frag32 qf[2];
    {
        int row = wave * 16 + mg;
        #pragma unroll
        for (int ks = 0; ks < 2; ++ks) {
            qf[ks].q[0] = *(const uint4*)&Qs[row][ks * 32 + g * 8];
            qf[ks].q[1] = *(const uint4*)&Qs[row][ks * 32 + 16 + g * 8];
        }
    }

    v8f zero = {0,0,0,0,0,0,0,0};
    v8f o[4];
    float m[8], l[8];
    #pragma unroll
    for (int tj = 0; tj < 4; ++tj) o[tj] = zero;
    #pragma unroll
    for (int r = 0; r < 8; ++r) { m[r] = -3.0e38f; l[r] = 0.f; }

    const float norm = 0.125f;   // 1/sqrt(64)

    for (int kb = 0; kb < S / 64; ++kb) {
        __syncthreads();
        const int kv0 = kb * 64;

        // ---- K tile (64x64) via Tensor Data Mover: rows 128B + 16B pad ----
#if defined(__gfx1250__)
        if (wave == 0)
            tdm_load_2d_bf16(lds_addr_of(&Ksh[0][0]),
                             base + (size_t)kv0 * C3 + E + h * HD,
                             /*tile_x=*/64, /*tile_y=*/64,
                             /*row_stride=*/C3,
                             /*pad_interval: 32 dwords*/4, /*pad: 4 dwords*/3);
#endif

        // ---- V tile (64x64) transposed into LDS via VGPRs ----
        #pragma unroll
        for (int j = 0; j < 2; ++j) {
            int lin = tid + j * 256;             // 512 chunks of 8
            int r = lin >> 3, c8 = lin & 7;
#if !defined(__gfx1250__)
            uint4 dk = *(const uint4*)(base + (size_t)(kv0 + r) * C3 + E + h * HD + c8 * 8);
            *(uint4*)&Ksh[r][c8 * 8] = dk;
#endif
            uint4 dv = *(const uint4*)(base + (size_t)(kv0 + r) * C3 + 2 * E + h * HD + c8 * 8);
            const unsigned short* pv = (const unsigned short*)&dv;
            #pragma unroll
            for (int t = 0; t < 8; ++t) Vt[c8 * 8 + t][r] = pv[t];
        }
#if defined(__gfx1250__)
        if (wave == 0) __builtin_amdgcn_s_wait_tensorcnt(0);
#endif
        __syncthreads();

        // ---- S = (Q K^T) : 4 col-tiles x 2 k-steps ----
        v8f s[4];
        #pragma unroll
        for (int tj = 0; tj < 4; ++tj) {
            s[tj] = zero;
            int kr = tj * 16 + mg;
            #pragma unroll
            for (int ks = 0; ks < 2; ++ks) {
                Frag32 bk;
                bk.q[0] = *(const uint4*)&Ksh[kr][ks * 32 + g * 8];
                bk.q[1] = *(const uint4*)&Ksh[kr][ks * 32 + 16 + g * 8];
                s[tj] = wmma_bf16(qf[ks], bk, s[tj]);
            }
        }

        // ---- online softmax (rows = r + 8*g, cols striped over 16 lanes) ----
        #pragma unroll
        for (int r = 0; r < 8; ++r) {
            float mx = -3.0e38f;
            #pragma unroll
            for (int tj = 0; tj < 4; ++tj) {
                s[tj][r] *= norm;
                mx = fmaxf(mx, s[tj][r]);
            }
            #pragma unroll
            for (int off = 1; off < 16; off <<= 1)
                mx = fmaxf(mx, __shfl_xor(mx, off, 32));
            float mn   = fmaxf(m[r], mx);
            float corr = __expf(m[r] - mn);
            m[r] = mn;
            float rs = 0.f;
            #pragma unroll
            for (int tj = 0; tj < 4; ++tj) {
                float e = __expf(s[tj][r] - mn);
                rs += e;
                Ps[wave][r + 8 * g][tj * 16 + mg] = f2bf(e);
            }
            #pragma unroll
            for (int off = 1; off < 16; off <<= 1)
                rs += __shfl_xor(rs, off, 32);
            l[r] = l[r] * corr + rs;
            #pragma unroll
            for (int tj = 0; tj < 4; ++tj) o[tj][r] *= corr;
        }

        // ---- O += P @ V  (A from per-wave Ps, B from transposed Vt) ----
        #pragma unroll
        for (int ks = 0; ks < 2; ++ks) {
            Frag32 pa;
            pa.q[0] = *(const uint4*)&Ps[wave][mg][ks * 32 + g * 8];
            pa.q[1] = *(const uint4*)&Ps[wave][mg][ks * 32 + 16 + g * 8];
            #pragma unroll
            for (int tj = 0; tj < 4; ++tj) {
                Frag32 bv;
                int dr = tj * 16 + mg;
                bv.q[0] = *(const uint4*)&Vt[dr][ks * 32 + g * 8];
                bv.q[1] = *(const uint4*)&Vt[dr][ks * 32 + 16 + g * 8];
                o[tj] = wmma_bf16(pa, bv, o[tj]);
            }
        }
    }

    // ---- normalize and store O (bf16, [b, s, 1024], head at col h*64) ----
    #pragma unroll
    for (int tj = 0; tj < 4; ++tj) {
        #pragma unroll
        for (int r = 0; r < 8; ++r) {
            int row = q0 + wave * 16 + r + 8 * g;
            int col = h * HD + tj * 16 + mg;
            float v = o[tj][r] / l[r];
            Obuf[(size_t)((size_t)b * S + row) * E + col] = f2bf(v);
        }
    }
}

// ---------------------------------------------------------------------------
extern "C" void kernel_launch(void* const* d_in, const int* in_sizes, int n_in,
                              void* d_out, int out_size, void* d_ws, size_t ws_size,
                              hipStream_t stream) {
    (void)in_sizes; (void)n_in; (void)out_size; (void)ws_size;

    const float* x     = (const float*)d_in[0];   // [4,2048,1024]
    const float* W_qkv = (const float*)d_in[1];   // [1024,3072]
    const float* b_qkv = (const float*)d_in[2];   // [3072]
    const float* W_out = (const float*)d_in[3];   // [1024,1024]
    const float* b_out = (const float*)d_in[4];   // [1024]

    const int M = 4 * 2048;       // 8192 tokens

    unsigned short* qkv = (unsigned short*)d_ws;             // [8192,3072] bf16 (48 MB)
    unsigned short* Ob  = qkv + (size_t)M * 3072;            // [8192,1024] bf16 (16 MB)

    // 1) QKV = x @ W_qkv + b_qkv  -> bf16
    gemm_bias_kernel<false, true><<<dim3(3072 / 128, M / 128), 256, 0, stream>>>(
        x, W_qkv, b_qkv, qkv, M, 3072, 1024);

    // 2) flash attention per (batch, head), 128 query rows per block
    attn_kernel<<<dim3(2048 / 128, 4 * 16), 256, 0, stream>>>(qkv, Ob);

    // 3) out = O @ W_out + b_out -> f32
    gemm_bias_kernel<true, false><<<dim3(1024 / 128, M / 128), 256, 0, stream>>>(
        Ob, W_out, b_out, d_out, M, 1024, 1024);
}
